// PriorCHMM_68247030334404
// MI455X (gfx1250) — matrix operand order repaired and checked by reading, not compiled, not run on hardware
//
#include <hip/hip_runtime.h>
#include <hip/hip_bf16.h>
#include <math.h>

// ---------------- problem constants ----------------
#define Bn      128
#define Sn      512
#define D_EMB   768
#define N_SRC   16
#define N_ENT   18
#define D_OBS   37              // 2*N_ENT + 1
#define NCOL    1369            // D_OBS*D_OBS
#define NTILES  88              // padded so every wave owns exactly 11 tiles
#define TPW     11              // tiles per wave (88 / 8 waves)
#define NPAD    (NTILES * 16)   // 1408 packed-W columns
#define SROW    1412            // LDS row stride (floats): %64 == 4 -> no bank clash
#define KSTEPS  24              // 768 / 32
#define MTOK    (Bn * Sn)       // 65536 tokens

// d_out layout (floats), in reference return order
#define OFF_TRANS  0
#define OFF_DIR    ((size_t)Bn * Sn * NCOL)                  // 89,718,784
#define OFF_O2O    (OFF_DIR + (size_t)Bn * N_SRC * NCOL)     // + 2,803,712
#define OFF_E2E    (OFF_O2O + (size_t)Bn * N_SRC)            // + 2048

typedef __attribute__((ext_vector_type(16))) __bf16 v16bf;
typedef __attribute__((ext_vector_type(8)))  float  v8f;

// ---------------------------------------------------------------
// Kernel 1: W_trans (768 x 1369 f32 row-major) -> Wt (1408 x 768 bf16,
// N-major). Columns n >= 1369 are zero, so the GEMM hot loop is guard-free.
// ---------------------------------------------------------------
__global__ void wprep_kernel(const float* __restrict__ W, __bf16* __restrict__ Wt) {
    int idx = blockIdx.x * 256 + threadIdx.x;
    if (idx >= NPAD * D_EMB) return;
    int n = idx / D_EMB;
    int k = idx - n * D_EMB;
    float v = (n < NCOL) ? W[(size_t)k * NCOL + n] : 0.0f;
    Wt[idx] = (__bf16)v;
}

// ---------------------------------------------------------------
// Kernel 2: fused GEMM (bf16 WMMA, f32 accum) + bias + row-softmax(37).
// One block = 16 tokens, 8 waves x 11 N-tiles, software-pipelined A/B
// fragments, logits staged in 16 x 1412 f32 dynamic LDS.
// ---------------------------------------------------------------
__global__ void gemm_softmax_kernel(const float* __restrict__ embs,
                                    const __bf16* __restrict__ Wt,
                                    const float* __restrict__ b_trans,
                                    float* __restrict__ out) {
    extern __shared__ float smem[];                 // 16 * SROW floats

    const int tid  = threadIdx.x;
    const int wave = tid >> 5;                      // 0..7
    const int lane = tid & 31;
    const int half = lane >> 4;                     // 0 or 1
    const int c    = lane & 15;                     // A row / B col / C col
    const int t0   = blockIdx.x * 16;               // first token of block

    // wave index as a provable SGPR -> scalar B base, VGPR carries only the
    // small per-lane component (kills 64-bit vector address math + WAR NOPs)
    const int wv = __builtin_amdgcn_readfirstlane(wave);

    v8f acc[TPW];
    #pragma unroll
    for (int t = 0; t < TPW; ++t) acc[t] = (v8f){0, 0, 0, 0, 0, 0, 0, 0};

    const float*  arow  = embs + (size_t)(t0 + c) * D_EMB + half * 16;
    const __bf16* bbase = Wt + (size_t)wv * (16 * D_EMB);   // scalar part
    const int     blane = c * D_EMB + half * 16;            // per-lane elements

    // A fragment (16x32 bf16), contiguous layout matching B:
    // lane<16 : K = 0..15; lane>=16 : K = 16..31
    auto loadA = [&](int k0) {
        const float* ap = arow + k0;
        v16bf a;
        #pragma unroll
        for (int e = 0; e < 8; ++e) a[e]     = (__bf16)ap[e];
        #pragma unroll
        for (int e = 0; e < 8; ++e) a[8 + e] = (__bf16)ap[8 + e];
        return a;
    };
    // B fragment (32x16 bf16): lane<16 holds col n0+c K=k0..k0+15,
    // lane>=16 same col K=k0+16..k0+31; contiguous 32 B in packed Wt.
    auto loadB = [&](int t, int k0) {
        return *(const v16bf*)(bbase + (size_t)t * (8 * 16 * D_EMB) + k0 + blane);
    };

    v16bf a = loadA(0);
    v16bf b = loadB(0, 0);

    for (int kb = 0; kb < KSTEPS; ++kb) {
        const int k0 = kb * 32;
        // prefetch next A early: its loads + cvt_pk_bf16 fill WMMA hazard slots
        v16bf anext = (kb + 1 < KSTEPS) ? loadA(k0 + 32) : a;

        #pragma unroll
        for (int t = 0; t < TPW; ++t) {
            v16bf bnext = (t + 1 < TPW) ? loadB(t + 1, k0)
                        : ((kb + 1 < KSTEPS) ? loadB(0, k0 + 32) : b);
            acc[t] = __builtin_amdgcn_wmma_f32_16x16x32_bf16(
                false, a, false, b, (short)0, acc[t], false, false);
            b = bnext;
        }
        a = anext;
    }

    // ---- spill C tiles to LDS: row = i + half*8, col = n0 + c
    #pragma unroll
    for (int t = 0; t < TPW; ++t) {
        const int n0 = (wave + t * 8) * 16;
        #pragma unroll
        for (int i = 0; i < 8; ++i)
            smem[(i + half * 8) * SROW + n0 + c] = acc[t][i];
    }

    __syncthreads();

    // ---- bias + softmax over 37-wide groups: 16 tokens * 37 rows = 592
    for (int g = tid; g < 16 * D_OBS; g += 256) {
        const int tt = g / D_OBS;
        const int h  = g - tt * D_OBS;
        const float* rowp = smem + tt * SROW + h * D_OBS;
        const float* bp   = b_trans + h * D_OBS;

        float m = -1e30f;
        #pragma unroll
        for (int j = 0; j < D_OBS; ++j) m = fmaxf(m, rowp[j] + bp[j]);

        float s = 0.0f;
        #pragma unroll
        for (int j = 0; j < D_OBS; ++j) s += __expf(rowp[j] + bp[j] - m);

        const float inv = 1.0f / s;
        float* op = out + OFF_TRANS + (size_t)(t0 + tt) * NCOL + h * D_OBS;
        #pragma unroll
        for (int j = 0; j < D_OBS; ++j) op[j] = __expf(rowp[j] + bp[j] - m) * inv;
    }
}

// ---------------------------------------------------------------
// Kernel 3: head on e0 = embs[:,0,:].
// tid<16 -> conc_o2o = sigmoid(e0 @ W_2o + b_2o)
// else   -> conc_e2e = prob_scaling(sigmoid(e0 @ W_2e + b_2e))
// ---------------------------------------------------------------
__global__ void head_kernel(const float* __restrict__ embs,
                            const float* __restrict__ W2o, const float* __restrict__ b2o,
                            const float* __restrict__ W2e, const float* __restrict__ b2e,
                            float* __restrict__ out) {
    const int b   = blockIdx.x;
    const int tid = threadIdx.x;
    if (tid >= N_SRC + N_SRC * N_ENT) return;       // 304 workers
    const float* e0 = embs + (size_t)b * Sn * D_EMB;

    if (tid < N_SRC) {
        float acc = b2o[tid];
        for (int k = 0; k < D_EMB; ++k) acc = fmaf(e0[k], W2o[k * N_SRC + tid], acc);
        out[OFF_O2O + b * N_SRC + tid] = 1.0f / (1.0f + __expf(-acc));
    } else {
        const int j = tid - N_SRC;                  // 0..287
        float acc = b2e[j];
        for (int k = 0; k < D_EMB; ++k) acc = fmaf(e0[k], W2e[k * (N_SRC * N_ENT) + j], acc);
        float x = 1.0f / (1.0f + __expf(-acc));
        // prob_scaling(x, p=1/16, t1=2, t2=4)
        const float p = 1.0f / 16.0f;
        float y;
        if (x < p) {
            float u = x / p;
            y = u * u * p;
        } else {
            float u  = (1.0f - x) / (1.0f - p);
            float u2 = u * u;
            y = 1.0f - u2 * u2 * (1.0f - p);
        }
        out[OFF_E2E + b * (N_SRC * N_ENT) + j] = y;
    }
}

// ---------------------------------------------------------------
// Kernel 4: build Dirichlet concentration rows, emit normalized mean
// (deterministic stand-in for the Dirichlet sample). One thread per
// (batch, src, row): 128*16*37 rows of 37.
// ---------------------------------------------------------------
__global__ void emiss_kernel(const float* __restrict__ conc_o2o,
                             const float* __restrict__ conc_e2e,
                             float* __restrict__ dir) {
    const int idx = blockIdx.x * 256 + threadIdx.x;
    if (idx >= Bn * N_SRC * D_OBS) return;
    const int r  = idx % D_OBS;
    const int bs = idx / D_OBS;

    float row[D_OBS];
    if (r == 0) {
        const float cv = conc_o2o[bs];
        row[0] = cv;
        const float rest = (1.0f - cv) / (float)(D_OBS - 1);
        #pragma unroll
        for (int j = 1; j < D_OBS; ++j) row[j] = rest;
    } else {
        const int ent = (r - 1) >> 1;
        const float x = conc_e2e[bs * N_ENT + ent];
        const float tp   = 0.5f / (float)D_OBS;
        const float y_tp = 1.0f - tp * (float)(D_OBS - 1);
        const float u    = (1.0f - x) / (1.0f - tp);
        const float o    = (x <= tp) ? (1.0f - (float)(D_OBS - 1) * x) : y_tp * u * u;
        const float nd   = (1.0f - x - o) / (float)(D_OBS - 2);
        row[0] = o;
        #pragma unroll
        for (int j = 1; j < D_OBS; ++j) row[j] = (j == r) ? x : nd;
    }

    float sum = 0.0f;
    #pragma unroll
    for (int j = 0; j < D_OBS; ++j) {
        const float a = row[j] * 99.0f + 1.0f;      // alpha = v*(CONC_MAX-CONC_BASE)+CONC_BASE
        row[j] = a;
        sum += a;
    }
    const float inv = 1.0f / sum;
    float* op = dir + (size_t)idx * D_OBS;
    #pragma unroll
    for (int j = 0; j < D_OBS; ++j) op[j] = row[j] * inv;
}

// ---------------------------------------------------------------
extern "C" void kernel_launch(void* const* d_in, const int* in_sizes, int n_in,
                              void* d_out, int out_size, void* d_ws, size_t ws_size,
                              hipStream_t stream) {
    const float* embs    = (const float*)d_in[0];
    const float* W_trans = (const float*)d_in[1];
    const float* b_trans = (const float*)d_in[2];
    const float* W_2o    = (const float*)d_in[3];
    const float* b_2o    = (const float*)d_in[4];
    const float* W_2e    = (const float*)d_in[5];
    const float* b_2e    = (const float*)d_in[6];
    float* out = (float*)d_out;
    __bf16* Wt = (__bf16*)d_ws;                     // 1408*768*2 B = 2.16 MB

    // 1) pack W_trans -> bf16, N-major, zero-padded to 1408 cols
    {
        const int n = NPAD * D_EMB;
        wprep_kernel<<<(n + 255) / 256, 256, 0, stream>>>(W_trans, Wt);
    }

    // 2) fused GEMM + softmax: one block per 16 tokens, ~88 KB dynamic LDS
    {
        const size_t lds = (size_t)16 * SROW * sizeof(float);
        gemm_softmax_kernel<<<MTOK / 16, 256, lds, stream>>>(embs, Wt, b_trans, out);
    }

    // 3) head (conc_o2o, conc_e2e)
    head_kernel<<<Bn, 320, 0, stream>>>(embs, W_2o, b_2o, W_2e, b_2e, out);

    // 4) emission matrices (Dirichlet mean)
    {
        const int n = Bn * N_SRC * D_OBS;
        emiss_kernel<<<(n + 255) / 256, 256, 0, stream>>>(
            out + OFF_O2O, out + OFF_E2E, out + OFF_DIR);
    }
}
